// AttentivePoolingNetwork_39247411150857
// MI455X (gfx1250) — matrix operand order repaired and checked by reading, not compiled, not run on hardware
//
#include <hip/hip_runtime.h>
#include <hip/hip_bf16.h>
#include <math.h>

typedef __attribute__((ext_vector_type(16))) _Float16 v16h;
typedef __attribute__((ext_vector_type(8)))  float    v8f;

#define B_   64
#define LS_  128
#define N_   64
#define LC_  32
#define D_   128
#define F_   128
#define NEG_INF_ (-1e9f)

// Fragment-tile size in halves: 32 lanes * 16 halves
#define FRAG_ (32 * 16)

// ---------------- CDNA5 async global->LDS (ASYNCcnt-tracked DMA per lane) ---------------
// LDS address taken from the VDST VGPR operand; global addr from VGPR pair; SADDR=off (GV mode).
__device__ __forceinline__ void async_g2lds_b128(unsigned int lds_off, const void* gptr) {
  asm volatile("global_load_async_to_lds_b128 %0, %1, off"
               :: "v"(lds_off), "v"(gptr) : "memory");
}
__device__ __forceinline__ void async_wait0() {
  asm volatile("s_wait_asynccnt 0x0" ::: "memory");
}

// ---------------- WMMA fragment loaders (wave32, v_wmma_f32_16x16x32_f16) ---------------
// A-matrix 16x32 (MxK), 16 halves/lane.  Contiguous-K layout used here (B matches):
//   lanes 0-15 : M=lane,    halves 0-15 -> K=0..15
//   lanes 16-31: M=lane-16, halves 0-15 -> K=16..31
__device__ __forceinline__ v16h load_A_h(const _Float16* src, int ld) {
  const int lane = threadIdx.x & 31;
  const int m  = lane & 15;
  const int k0 = (lane >> 4) * 16;
  const int k1 = k0 + 8;
  v16h a;
#pragma unroll
  for (int h = 0; h < 8; ++h) a[h]   = src[m * ld + k0 + h];
#pragma unroll
  for (int h = 0; h < 8; ++h) a[8+h] = src[m * ld + k1 + h];
  return a;
}

// B-matrix 32x16 (KxN) from an N-major (row-major N x K) source: element (k,n)=src[n*ld+k].
//   lanes 0-15 : N=lane,    half h -> K=h ; lanes 16-31: N=lane-16, half h -> K=16+h
__device__ __forceinline__ v16h load_Bt_h(const _Float16* src, int ld) {
  const int lane = threadIdx.x & 31;
  const int n  = lane & 15;
  const int kb = (lane >> 4) * 16;
  v16h b;
#pragma unroll
  for (int h = 0; h < 16; ++h) b[h] = src[n * ld + kb + h];
  return b;
}

// Pre-packed fragment: per-lane 16 contiguous halves (32B aligned) -> 2x b128 loads.
__device__ __forceinline__ v16h load_frag_pack(const _Float16* tile) {
  const int lane = threadIdx.x & 31;
  return *(const v16h*)(tile + lane * 16);
}

#define WMMA_F16(A, Bm, C) \
  __builtin_amdgcn_wmma_f32_16x16x32_f16(false, (A), false, (Bm), (short)0, (C), false, false)

// ============ Kernel 0: pack the three 128x128 f32 weights into A-fragment f16 ==========
// Layout: wpack[wi][tile = mt*4+kk][lane][h], tile = 512 halves. One wave per tile.
__global__ void __launch_bounds__(32)
apn_pack_weights(const float* __restrict__ w0, const float* __restrict__ w1,
                 const float* __restrict__ w2, _Float16* __restrict__ wpack) {
  const int wi = blockIdx.x >> 5;     // matrix 0..2
  const int t  = blockIdx.x & 31;     // tile 0..31 (mt*4 + kk)
  const float* w = (wi == 0) ? w0 : ((wi == 1) ? w1 : w2);
  const int mt = t >> 2, kk = t & 3;
  const int lane = threadIdx.x;
  const int m  = mt * 16 + (lane & 15);
  const int k0 = (lane >> 4) * 16;
  const int k1 = k0 + 8;
  v16h a;
#pragma unroll
  for (int h = 0; h < 8; ++h) a[h]   = (_Float16)w[m * 128 + kk * 32 + k0 + h];
#pragma unroll
  for (int h = 0; h < 8; ++h) a[8+h] = (_Float16)w[m * 128 + kk * 32 + k1 + h];
  *(v16h*)(wpack + ((wi * 32 + t) * 32 + lane) * 16) = a;
}

// ======================= Kernel 1: conv_s per batch (shared over N claims) ==============
// conv_s[b,f,s] = sum_d sent_w[f,d] * emb[sentences[b,s], d] + sent_b[f]
__global__ void __launch_bounds__(128)
apn_conv_s(const int* __restrict__ sentences, const float* __restrict__ emb,
           const _Float16* __restrict__ wSentPack, const float* __restrict__ sent_b,
           float* __restrict__ convs32) {
  __shared__ _Float16 sE[LS_ * D_];   // gathered sentence embeddings, [s][d], 32 KB

  const int b = blockIdx.x, tid = threadIdx.x;
  const int wave = tid >> 5, lane = tid & 31;
  const int lm = lane & 15, lhi = lane >> 4;

  for (int idx = tid; idx < LS_ * D_; idx += 128) {
    const int s = idx >> 7, d = idx & 127;
    const int tok = sentences[b * LS_ + s];
    sE[idx] = (_Float16)emb[tok * D_ + d];
  }
  __syncthreads();

  for (int t = wave; t < 64; t += 4) {
    const int mt = t >> 3, nt = t & 7;
    v8f acc = {};
#pragma unroll
    for (int kk = 0; kk < 4; ++kk) {
      v16h a  = load_frag_pack(wSentPack + (mt * 4 + kk) * FRAG_);
      v16h bm = load_Bt_h(sE + nt * 16 * D_ + kk * 32, D_);
      acc = WMMA_F16(a, bm, acc);
    }
    const int m0 = mt * 16 + lhi * 8;
    const int c  = nt * 16 + lm;
#pragma unroll
    for (int r = 0; r < 8; ++r) {
      const int m = m0 + r;
      convs32[b * (F_ * LS_) + m * LS_ + c] = acc[r] + sent_b[m];
    }
  }
}

// ====== Kernel 1b: repack conv_s(b) into GEMM3 B-fragment layout (f16), once per b ======
// cspack[b][tile = kk*8+nt][lane][h] where (lane,h): n=lane&15, k=(lane>>4)*16+h.
__global__ void __launch_bounds__(128)
apn_pack_convs(const float* __restrict__ convs32, _Float16* __restrict__ cspack) {
  __shared__ _Float16 sC[F_ * LS_];   // conv_s f16 [f][s], 32 KB
  const int b = blockIdx.x, tid = threadIdx.x;
  for (int idx = tid; idx < F_ * LS_; idx += 128)
    sC[idx] = (_Float16)convs32[b * (F_ * LS_) + idx];
  __syncthreads();

  const int wave = tid >> 5, lane = tid & 31;
  const int n = lane & 15, kb = (lane >> 4) * 16;
  for (int t = wave; t < 32; t += 4) {
    const int kk = t >> 3, nt = t & 7;
    v16h bf;
#pragma unroll
    for (int h = 0; h < 16; ++h)
      bf[h] = sC[(kk * 32 + kb + h) * LS_ + nt * 16 + n];
    *(v16h*)(cspack + (b * 32 + t) * FRAG_ + lane * 16) = bf;
  }
}

// ============ Kernel 2: fully fused per-(b,n) pipeline, nothing big hits HBM ============
__global__ void __launch_bounds__(128)
apn_claim_fused(const int* __restrict__ claims, const int* __restrict__ smask,
                const int* __restrict__ cmask, const float* __restrict__ emb,
                const _Float16* __restrict__ wClaimPack, const float* __restrict__ claim_b,
                const _Float16* __restrict__ wFcPack, const float* __restrict__ fc_b,
                const float* __restrict__ convs32, const _Float16* __restrict__ cspack,
                float* __restrict__ out) {
  __shared__ _Float16 sE[LC_ * D_];     // emb_c  [l][d]         8 KB
  __shared__ float    sCc[F_ * LC_];    // conv_c [f][l] f32    16 KB (encoder sum)
  __shared__ _Float16 sCcT16[LC_ * F_]; // conv_c [l][f] f16     8 KB (GEMM2 B, N-major)
  __shared__ _Float16 sLhs[LC_ * F_];   // lhs    [l][g] f16     8 KB (GEMM3 A)
  // sT doubles as the async-DMA landing zone for the raw f32 claim embeddings
  // (LC_*D_ f32 == LC_*LS_ f32 == 16 KB), dead until GEMM3 overwrites it.
  __shared__ float    sT[LC_ * LS_];    // transformed [l][s]   16 KB
  __shared__ float    sAs[LS_];
  __shared__ float    sAc[LC_];
  __shared__ float    sEs[F_];
  __shared__ float    sEc[F_];

  const int bn = blockIdx.x;
  const int b = bn >> 6, n = bn & (N_ - 1);
  const int tid = threadIdx.x;
  const int wave = tid >> 5, lane = tid & 31;
  const int lm = lane & 15, lhi = lane >> 4;

  const _Float16* csp = cspack + b * (32 * FRAG_);
  __builtin_prefetch(csp + tid * LS_, 0, 1);   // warm this block's conv_s fragments

  // Async-DMA the claim-embedding rows (f32) straight into LDS (sT staging),
  // 16B per lane per op, no VGPR round-trip.  1024 chunks / 128 threads = 8 ops each.
#pragma unroll
  for (int c = tid; c < (LC_ * D_) / 4; c += 128) {
    const int l = c >> 5;            // 32 x 16B chunks per 512B embedding row
    const int q = c & 31;
    const int tok = claims[(b * N_ + n) * LC_ + l];
    async_g2lds_b128((unsigned int)(uintptr_t)&sT[c * 4],
                     emb + (size_t)tok * D_ + q * 4);
  }
  async_wait0();
  __syncthreads();

  // Convert staged f32 -> f16 operand tile [l][d]
  for (int idx = tid; idx < LC_ * D_; idx += 128)
    sE[idx] = (_Float16)sT[idx];
  __syncthreads();

  // GEMM1: conv_c[f][l] = claim_w(F,D) * emb_c^T(D,Lc) + claim_b.  M=128,N=32,K=128.
  for (int t = wave; t < 16; t += 4) {
    const int mt = t >> 1, nt = t & 1;
    v8f acc = {};
#pragma unroll
    for (int kk = 0; kk < 4; ++kk) {
      v16h a  = load_frag_pack(wClaimPack + (mt * 4 + kk) * FRAG_);
      v16h bm = load_Bt_h(sE + nt * 16 * D_ + kk * 32, D_);
      acc = WMMA_F16(a, bm, acc);
    }
    const int m0 = mt * 16 + lhi * 8;
    const int c  = nt * 16 + lm;
#pragma unroll
    for (int r = 0; r < 8; ++r) {
      const int m = m0 + r;
      const float v = acc[r] + claim_b[m];
      sCc[m * LC_ + c]    = v;            // [f][l] f32
      sCcT16[c * F_ + m]  = (_Float16)v;  // [l][f] f16 (N-major for GEMM2 B)
    }
  }
  __syncthreads();

  // GEMM2: lhs^T[g][l] = fc_w(F,F) * conv_c16(F,Lc) + fc_b; stored transposed as sLhs[l][g].
  for (int t = wave; t < 16; t += 4) {
    const int mt = t >> 1, nt = t & 1;
    v8f acc = {};
#pragma unroll
    for (int kk = 0; kk < 4; ++kk) {
      v16h a  = load_frag_pack(wFcPack + (mt * 4 + kk) * FRAG_);
      v16h bm = load_Bt_h(sCcT16 + nt * 16 * F_ + kk * 32, F_);
      acc = WMMA_F16(a, bm, acc);
    }
    const int m0 = mt * 16 + lhi * 8;
    const int c  = nt * 16 + lm;
#pragma unroll
    for (int r = 0; r < 8; ++r) {
      const int g = m0 + r;
      sLhs[c * F_ + g] = (_Float16)(acc[r] + fc_b[g]);
    }
  }
  __syncthreads();

  // GEMM3 (dominant): T[l][s] = tanh( lhs(Lc,F) * conv_s16(F,Ls) ).  M=32,N=128,K=128.
  for (int t = wave; t < 16; t += 4) {
    const int mt = t >> 3, nt = t & 7;
    v8f acc = {};
#pragma unroll
    for (int kk = 0; kk < 4; ++kk) {
      v16h a  = load_A_h(sLhs + mt * 16 * F_ + kk * 32, F_);
      v16h bm = load_frag_pack(csp + (kk * 8 + nt) * FRAG_);
      acc = WMMA_F16(a, bm, acc);
    }
    const int m0 = mt * 16 + lhi * 8;
    const int c  = nt * 16 + lm;
#pragma unroll
    for (int r = 0; r < 8; ++r)
      sT[(m0 + r) * LS_ + c] = tanhf(acc[r]);
  }
  __syncthreads();

  // ---- attn over sentence tokens: softmax_s( max_l T[l][s] ), masked. tid == s. ----
  const float msk_s = (float)smask[b * LS_ + tid];
  float pooled_s = sT[tid];
  for (int l = 1; l < LC_; ++l) pooled_s = fmaxf(pooled_s, sT[l * LS_ + tid]);
  const float logit_s = (msk_s > 0.f) ? pooled_s : NEG_INF_;
  sAs[tid] = logit_s;
  __syncthreads();
  float mxs = -3.4e38f;
  for (int i = 0; i < LS_; ++i) mxs = fmaxf(mxs, sAs[i]);
  const float es = (msk_s > 0.f) ? __expf(logit_s - mxs) : 0.f;
  __syncthreads();
  sAs[tid] = es;
  __syncthreads();
  float sums = 0.f;
  for (int i = 0; i < LS_; ++i) sums += sAs[i];
  __syncthreads();
  sAs[tid] = es / sums;

  // ---- attn over claim tokens: softmax_l( max_s T[l][s] ), masked. tid<32 == l. ----
  float logit_c = NEG_INF_, msk_c = 0.f;
  if (tid < LC_) {
    msk_c = (float)cmask[(b * N_ + n) * LC_ + tid];
    float pc = sT[tid * LS_];
    for (int s = 1; s < LS_; ++s) pc = fmaxf(pc, sT[tid * LS_ + s]);
    logit_c = (msk_c > 0.f) ? pc : NEG_INF_;
    sAc[tid] = logit_c;
  }
  __syncthreads();
  float mxc = -3.4e38f;
  for (int i = 0; i < LC_; ++i) mxc = fmaxf(mxc, sAc[i]);
  const float ec = (tid < LC_ && msk_c > 0.f) ? __expf(logit_c - mxc) : 0.f;
  __syncthreads();
  if (tid < LC_) sAc[tid] = ec;
  __syncthreads();
  float sumc = 0.f;
  for (int i = 0; i < LC_; ++i) sumc += sAc[i];
  __syncthreads();
  if (tid < LC_) sAc[tid] = ec / sumc;
  __syncthreads();

  // ---- encoders (f32, matching reference precision path). tid == feature f. ----
  const float* cs32 = convs32 + b * (F_ * LS_) + tid * LS_;
  float encs = 0.f;
  for (int s = 0; s < LS_; ++s) encs += sAs[s] * cs32[s];
  float encc = 0.f;
  for (int l = 0; l < LC_; ++l) encc += sAc[l] * sCc[tid * LC_ + l];
  sEs[tid] = encs;
  sEc[tid] = encc;
  __syncthreads();

  if (tid == 0) {
    float dot = 0.f, ns = 0.f, nc = 0.f;
    for (int f = 0; f < F_; ++f) {
      dot += sEs[f] * sEc[f];
      ns  += sEs[f] * sEs[f];
      nc  += sEc[f] * sEc[f];
    }
    ns = fmaxf(sqrtf(ns), 1e-8f);
    nc = fmaxf(sqrtf(nc), 1e-8f);
    out[bn] = dot / (ns * nc);
  }
}

extern "C" void kernel_launch(void* const* d_in, const int* in_sizes, int n_in,
                              void* d_out, int out_size, void* d_ws, size_t ws_size,
                              hipStream_t stream) {
  const int*   sentences = (const int*)  d_in[0];
  const int*   smask     = (const int*)  d_in[1];
  const int*   claims    = (const int*)  d_in[2];
  const int*   cmask     = (const int*)  d_in[3];
  const float* emb       = (const float*)d_in[4];
  const float* sent_w    = (const float*)d_in[5];
  const float* sent_b    = (const float*)d_in[6];
  const float* claim_w   = (const float*)d_in[7];
  const float* claim_b   = (const float*)d_in[8];
  const float* fc_w      = (const float*)d_in[9];
  const float* fc_b      = (const float*)d_in[10];
  (void)in_sizes; (void)n_in; (void)out_size; (void)ws_size;

  float* out = (float*)d_out;                                   // (B, N) scores

  // Workspace: conv_s f32 (4 MB) | conv_s B-fragments f16 (2 MB) | packed weights (96 KB)
  float*    convs32 = (float*)d_ws;
  _Float16* cspack  = (_Float16*)(convs32 + B_ * F_ * LS_);
  _Float16* wpack   = cspack + B_ * 32 * FRAG_;
  _Float16* wSent   = wpack + 0 * 32 * FRAG_;
  _Float16* wClaim  = wpack + 1 * 32 * FRAG_;
  _Float16* wFc     = wpack + 2 * 32 * FRAG_;

  apn_pack_weights<<<96, 32, 0, stream>>>(sent_w, claim_w, fc_w, wpack);
  apn_conv_s<<<B_, 128, 0, stream>>>(sentences, emb, wSent, sent_b, convs32);
  apn_pack_convs<<<B_, 128, 0, stream>>>(convs32, cspack);
  apn_claim_fused<<<B_ * N_, 128, 0, stream>>>(claims, smask, cmask, emb,
                                               wClaim, claim_b, wFc, fc_b,
                                               convs32, cspack, out);
}
